// UltraNetBrevitasSimple_54992761258612
// MI455X (gfx1250) — 8-bit matrix operand order repaired and checked by reading, compiled, not run on hardware
//
#include <hip/hip_runtime.h>
#include <cstdint>
#include <cstddef>

typedef int v8i __attribute__((ext_vector_type(8)));

#define QEPS 1e-5f

__device__ __forceinline__ float clampf(float x, float lo, float hi) {
    return fminf(fmaxf(x, lo), hi);
}

__global__ __launch_bounds__(256) void prep_weights_kernel(
    const float* __restrict__ w, int cout, int cin, int taps, int ntiles,
    float* __restrict__ sw_out, int8_t* __restrict__ wpack)
{
    __shared__ float red[256];
    __shared__ float s_sh;
    const int tid = threadIdx.x;
    const int count = cout * cin * taps;

    float m = 0.f;
    for (int i = tid; i < count; i += 256) m = fmaxf(m, fabsf(w[i]));
    red[tid] = m;
    __syncthreads();
    for (int off = 128; off > 0; off >>= 1) {
        if (tid < off) red[tid] = fmaxf(red[tid], red[tid + off]);
        __syncthreads();
    }
    if (tid == 0) { s_sh = red[0] / 7.0f; sw_out[0] = s_sh; }
    __syncthreads();

    const float s = s_sh;
    const float inv_s = (s > 0.f) ? (1.0f / s) : 0.f;
    const int total = ntiles * taps * 32 * 32;
    for (int idx = tid; idx < total; idx += 256) {
        const int bytei = idx & 31;
        const int lane  = (idx >> 5) & 31;
        const int tap   = (idx >> 10) % taps;
        const int nt    = idx / (1024 * taps);
        const int j = bytei >> 2, b = bytei & 3;
        const int half = lane >> 4, n = lane & 15;
        const int K = (j >> 1) * 16 + half * 8 + (j & 1) * 4 + b;
        const int c = nt * 16 + n;
        float q = 0.f;
        if (c < cout && K < cin) {
            const float wv = w[(c * cin + K) * taps + tap];
            q = clampf(rintf(wv * inv_s), -7.f, 7.f);
        }
        wpack[idx] = (int8_t)q;
    }
}

__global__ void quant_input_kernel(const float* __restrict__ x,
                                   const float* __restrict__ scales,
                                   int8_t* __restrict__ out)
{
    const int Np = 16, C = 3, H = 320, W = 640;
    const int total = Np * C * H * W;
    int idx = blockIdx.x * blockDim.x + threadIdx.x;
    if (idx >= total) return;
    const int c  = idx % C;
    const int xw = (idx / C) % W;
    const int y  = (idx / (C * W)) % H;
    const int n  = idx / (C * W * H);
    const float v  = x[((n * C + c) * H + y) * W + xw];
    const float s0 = scales[0], s1 = scales[1];
    const float f0 = clampf(rintf(v / s0), -8.f, 7.f) * s0;
    const float q  = clampf(rintf(f0 / s1), -8.f, 7.f);
    out[idx] = (int8_t)q;
}

__global__ void maxpool2_kernel(const int8_t* __restrict__ in, int8_t* __restrict__ out,
                                int C, int Ho, int Wo, int total)
{
    int idx = blockIdx.x * blockDim.x + threadIdx.x;
    if (idx >= total) return;
    const int c = idx % C;
    const int x = (idx / C) % Wo;
    const int y = (idx / (C * Wo)) % Ho;
    const int n = idx / (C * Wo * Ho);
    const int Wi = Wo * 2, Hi = Ho * 2;
    const int8_t* p = in + (size_t)((n * Hi + y * 2) * Wi + x * 2) * C + c;
    int8_t a = p[0], b = p[C], d = p[(size_t)Wi * C], e = p[(size_t)Wi * C + C];
    int8_t mx = a > b ? a : b;
    mx = d > mx ? d : mx;
    mx = e > mx ? e : mx;
    out[idx] = mx;
}

template <int CIN, int COUT>
__global__ __launch_bounds__(256) void conv3x3_wmma_kernel(
    const int8_t* __restrict__ act,
    const int8_t* __restrict__ wpack,
    const float* __restrict__ sw,
    const float* __restrict__ bn,
    const float* __restrict__ scales,
    int si, int sr, int sn,
    int8_t* __restrict__ out,
    int H, int W)
{
    constexpr int NT = COUT / 16;
    __shared__ __align__(16) unsigned char smem[3 * 66 * 64];

    const int tid  = threadIdx.x;
    const int lane = tid & 31;
    const int wave = tid >> 5;
    const int n  = blockIdx.z;
    const int y  = blockIdx.y;
    const int x0 = blockIdx.x * 64;

    if constexpr (CIN >= 16) {
        constexpr int CH = CIN / 16;
        for (int i = tid; i < 3 * 66 * 4; i += 256) {
            const int s  = i >> 2;
            const int ch = i & 3;
            const int r  = s / 66, px = s % 66;
            const int iy = y + r - 1;
            const int ix = x0 + px - 1;
            const bool ok = (iy >= 0 && iy < H && ix >= 0 && ix < W);
            uint4* dst = (uint4*)(smem + s * 64) + ch;
            if (ok && ch < CH) {
                const unsigned ldsoff =
                    (unsigned)(uintptr_t)dst;
                const uint64_t gaddr = (uint64_t)(uintptr_t)
                    (act + (size_t)((n * H + iy) * (size_t)W + ix) * CIN + ch * 16);
                asm volatile("global_load_async_to_lds_b128 %0, %1, off"
                             :: "v"(ldsoff), "v"(gaddr)
                             : "memory");
            } else {
                const uint4 z = {0u, 0u, 0u, 0u};
                *dst = z;
            }
        }
        asm volatile("s_wait_asynccnt 0" ::: "memory");
    } else {
        for (int s = tid; s < 3 * 66; s += 256) {
            const int r  = s / 66, px = s % 66;
            const int iy = y + r - 1;
            const int ix = x0 + px - 1;
            const bool ok = (iy >= 0 && iy < H && ix >= 0 && ix < W);
            uint4* dst = (uint4*)(smem + s * 64);
            const uint4 z = {0u, 0u, 0u, 0u};
            dst[0] = z; dst[1] = z; dst[2] = z; dst[3] = z;
            if (ok) {
                const int8_t* src =
                    act + (size_t)((n * H + iy) * (size_t)W + ix) * CIN;
                unsigned char* d8 = (unsigned char*)dst;
                d8[0] = (unsigned char)src[0];
                d8[1] = (unsigned char)src[1];
                d8[2] = (unsigned char)src[2];
            }
        }
    }
    __syncthreads();

    const int half = lane >> 4;
    const int m    = lane & 15;

    for (int t = wave; t < 4 * NT; t += 8) {
        const int mt = t / NT, nt = t % NT;
        __builtin_prefetch(wpack + (size_t)(nt * 9) * 1024, 0, 1);

        v8i acc = {0, 0, 0, 0, 0, 0, 0, 0};
#pragma unroll
        for (int tap = 0; tap < 9; ++tap) {
            const int ky = tap / 3, kx = tap % 3;
            const int slot = ky * 66 + mt * 16 + m + kx;
            const unsigned char* base = smem + slot * 64 + half * 8;
            const int2 a0 = *(const int2*)(base);
            const int2 a1 = *(const int2*)(base + 16);
            const int2 a2 = *(const int2*)(base + 32);
            const int2 a3 = *(const int2*)(base + 48);
            v8i A;
            A[0] = a0.x; A[1] = a0.y; A[2] = a1.x; A[3] = a1.y;
            A[4] = a2.x; A[5] = a2.y; A[6] = a3.x; A[7] = a3.y;
            const v8i B = *(const v8i*)(wpack + ((size_t)(nt * 9 + tap) * 32 + lane) * 32);
            acc = __builtin_amdgcn_wmma_i32_16x16x64_iu8(
                 true, A,  true, B, acc,
                 false,  false);
        }

        const int c = nt * 16 + m;
        const float g  = bn[0 * COUT + c];
        const float bb = bn[1 * COUT + c];
        const float mu = bn[2 * COUT + c];
        const float vv = bn[3 * COUT + c];
        const float inv    = g / sqrtf(vv + QEPS);
        const float s_in   = scales[si];
        const float s_relu = scales[sr];
        const float s_next = scales[sn];
        const float alpha  = s_in * sw[0] * inv;
        const float beta   = bb - mu * inv;
#pragma unroll
        for (int r = 0; r < 8; ++r) {
            const int M  = r + half * 8;
            const int ix = x0 + mt * 16 + M;
            if (ix < W) {
                float yv = fmaxf((float)acc[r] * alpha + beta, 0.f);
                float rq = clampf(rintf(yv / s_relu), 0.f, 15.f);
                float qn = clampf(rintf(rq * s_relu / s_next), -8.f, 7.f);
                out[(size_t)((n * H + y) * (size_t)W + ix) * COUT + c] = (int8_t)qn;
            }
        }
    }
}

__global__ __launch_bounds__(256) void head1x1_wmma_kernel(
    const int8_t* __restrict__ act, const int8_t* __restrict__ wpack,
    const float* __restrict__ sw, const float* __restrict__ scales,
    const float* __restrict__ b9, float* __restrict__ out)
{
    const int lane = threadIdx.x & 31;
    const int wave = threadIdx.x >> 5;
    const int t = blockIdx.x * 8 + wave;
    if (t >= 800 * 3) return;
    const int mt = t / 3, nt = t % 3;
    const int half = lane >> 4, m = lane & 15;

    const int8_t* arow = act + (size_t)(mt * 16 + m) * 64 + half * 8;
    const int2 a0 = *(const int2*)(arow);
    const int2 a1 = *(const int2*)(arow + 16);
    const int2 a2 = *(const int2*)(arow + 32);
    const int2 a3 = *(const int2*)(arow + 48);
    v8i A;
    A[0] = a0.x; A[1] = a0.y; A[2] = a1.x; A[3] = a1.y;
    A[4] = a2.x; A[5] = a2.y; A[6] = a3.x; A[7] = a3.y;
    const v8i B = *(const v8i*)(wpack + ((size_t)nt * 32 + lane) * 32);

    v8i acc = {0, 0, 0, 0, 0, 0, 0, 0};
    acc = __builtin_amdgcn_wmma_i32_16x16x64_iu8(true, A, true, B, acc, false, false);

    const float s = scales[9] * sw[0];
    const int c = nt * 16 + m;
#pragma unroll
    for (int r = 0; r < 8; ++r) {
        const int M = r + half * 8;
        const int p = mt * 16 + M;
        const int n = p / 800, rem = p % 800;
        const int yy = rem / 40, xx = rem % 40;
        if (c < 36)
            out[((size_t)(n * 36 + c) * 20 + yy) * 40 + xx] = (float)acc[r] * s + b9[c];
    }
}

extern "C" void kernel_launch(void* const* d_in, const int* in_sizes, int n_in,
                              void* d_out, int out_size, void* d_ws, size_t ws_size,
                              hipStream_t stream)
{
    (void)in_sizes; (void)n_in; (void)out_size; (void)ws_size;

    const float* x = (const float*)d_in[0];
    const float* w[9];
    for (int i = 0; i < 9; ++i) w[i] = (const float*)d_in[1 + i];
    const float* b9 = (const float*)d_in[10];
    const float* bns[8];
    for (int i = 0; i < 8; ++i) bns[i] = (const float*)d_in[11 + i];
    const float* scales = (const float*)d_in[19];

    unsigned char* ws = (unsigned char*)d_ws;
    float* sw = (float*)ws;
    static const size_t wpoff[9] = {256, 9472, 27904, 64768, 101632,
                                    138496, 175360, 212224, 249088};
    int8_t* wp[9];
    for (int i = 0; i < 9; ++i) wp[i] = (int8_t*)(ws + wpoff[i]);
    int8_t* bufA = (int8_t*)(ws + (1 << 18));
    int8_t* bufB = bufA + (size_t)56 * 1024 * 1024;

    const int couts[9] = {16, 32, 64, 64, 64, 64, 64, 64, 36};
    const int cins[9]  = {3, 16, 32, 64, 64, 64, 64, 64, 64};
    const int tapss[9] = {9, 9, 9, 9, 9, 9, 9, 9, 1};
    const int ntls[9]  = {1, 2, 4, 4, 4, 4, 4, 4, 3};
    for (int l = 0; l < 9; ++l)
        prep_weights_kernel<<<1, 256, 0, stream>>>(w[l], couts[l], cins[l],
                                                   tapss[l], ntls[l], sw + l, wp[l]);

    {
        const int total = 16 * 3 * 320 * 640;
        quant_input_kernel<<<(total + 255) / 256, 256, 0, stream>>>(x, scales, bufA);
    }

    conv3x3_wmma_kernel<3, 16><<<dim3(10, 320, 16), 256, 0, stream>>>(
        bufA, wp[0], sw + 0, bns[0], scales, 1, 10, 2, bufB, 320, 640);
    { int total = 16 * 160 * 320 * 16;
      maxpool2_kernel<<<(total + 255) / 256, 256, 0, stream>>>(bufB, bufA, 16, 160, 320, total); }

    conv3x3_wmma_kernel<16, 32><<<dim3(5, 160, 16), 256, 0, stream>>>(
        bufA, wp[1], sw + 1, bns[1], scales, 2, 11, 3, bufB, 160, 320);
    { int total = 16 * 80 * 160 * 32;
      maxpool2_kernel<<<(total + 255) / 256, 256, 0, stream>>>(bufB, bufA, 32, 80, 160, total); }

    conv3x3_wmma_kernel<32, 64><<<dim3(3, 80, 16), 256, 0, stream>>>(
        bufA, wp[2], sw + 2, bns[2], scales, 3, 12, 4, bufB, 80, 160);
    { int total = 16 * 40 * 80 * 64;
      maxpool2_kernel<<<(total + 255) / 256, 256, 0, stream>>>(bufB, bufA, 64, 40, 80, total); }

    conv3x3_wmma_kernel<64, 64><<<dim3(2, 40, 16), 256, 0, stream>>>(
        bufA, wp[3], sw + 3, bns[3], scales, 4, 13, 5, bufB, 40, 80);
    { int total = 16 * 20 * 40 * 64;
      maxpool2_kernel<<<(total + 255) / 256, 256, 0, stream>>>(bufB, bufA, 64, 20, 40, total); }

    conv3x3_wmma_kernel<64, 64><<<dim3(1, 20, 16), 256, 0, stream>>>(
        bufA, wp[4], sw + 4, bns[4], scales, 5, 14, 6, bufB, 20, 40);
    conv3x3_wmma_kernel<64, 64><<<dim3(1, 20, 16), 256, 0, stream>>>(
        bufB, wp[5], sw + 5, bns[5], scales, 6, 15, 7, bufA, 20, 40);
    conv3x3_wmma_kernel<64, 64><<<dim3(1, 20, 16), 256, 0, stream>>>(
        bufA, wp[6], sw + 6, bns[6], scales, 7, 16, 8, bufB, 20, 40);
    conv3x3_wmma_kernel<64, 64><<<dim3(1, 20, 16), 256, 0, stream>>>(
        bufB, wp[7], sw + 7, bns[7], scales, 8, 17, 9, bufA, 20, 40);

    head1x1_wmma_kernel<<<300, 256, 0, stream>>>(bufA, wp[8], sw + 8, scales, b9,
                                                 (float*)d_out);
}
